// ClozeModel_68994354643535
// MI455X (gfx1250) — compile-verified, matrix operand order checked by reading, not run on hardware
//
#include <hip/hip_runtime.h>
#include <stdint.h>

// ---------------- model dimensions ----------------
#define Bb    64      // batch
#define Dd    253     // emb dim
#define WORDW 256     // D + NF
#define Ww    256     // W (read/key width)
#define Nm    128     // memory slots
#define Hh    512     // H
#define KA    1024    // A width = WORD + W + H? no: [xt(256)|r(256)|h(512)]
#define NG    2048    // 4*H
#define WIN   1026    // 4*W+2
#define WINP  1040    // padded to 65*16
#define LDA   1032    // LDS row stride (bf16) -> 16B lane stride, bank-conflict free
#define CLIPV 20.0f
#define EPSV  1e-6f

typedef __attribute__((ext_vector_type(16))) __bf16 v16bf;
typedef __attribute__((ext_vector_type(8)))  float  v8f;

union FragU { struct { uint4 lo, hi; } u; v16bf v; };

__device__ __forceinline__ uint16_t f2bfu(float f){
  uint32_t x = __builtin_bit_cast(uint32_t, f);
  x += 0x7FFFu + ((x >> 16) & 1u);           // round-to-nearest-even
  return (uint16_t)(x >> 16);
}
__device__ __forceinline__ __bf16 f2bf(float f){
  uint16_t u = f2bfu(f); return __builtin_bit_cast(__bf16, u);
}
__device__ __forceinline__ float sigf(float x){ return 1.0f / (1.0f + __expf(-x)); }
__device__ __forceinline__ float softplusf(float x){ return x > 20.f ? x : log1pf(__expf(x)); }

// A fragment: 16x32 bf16, rows in lanes 0..15, K split per ISA 16-bit A layout.
// LDS rows are 16B aligned (LDA*2 = 2064 bytes, %16==0); two ds_read_b128 per lane.
__device__ __forceinline__ v16bf loadA(const __bf16* A, int rt, int k0, int lane){
  int m  = rt * 16 + (lane & 15);
  int kh = (lane >> 4) << 3;                 // 0 or 8
  const __bf16* p = A + m * LDA + k0 + kh;
  FragU u;
  u.u.lo = *reinterpret_cast<const uint4*>(p);       // K = kh + 0..7
  u.u.hi = *reinterpret_cast<const uint4*>(p + 16);  // K = 16 + kh + 0..7
  return u.v;
}
// B fragment: pre-packed in exact register order; each lane reads 32 contiguous bytes.
__device__ __forceinline__ v16bf loadB(const uint16_t* P, int ct, int kt, int KT, int lane){
  const uint16_t* p = P + ((((size_t)ct * KT + kt) * 32 + lane) << 4);
  FragU u;
  u.u.lo = *reinterpret_cast<const uint4*>(p);
  u.u.hi = *reinterpret_cast<const uint4*>(p + 8);
  return u.v;
}

// Pack an element W[k][n] (bf16) into the 32x16 B-tile register layout.
__device__ __forceinline__ size_t bpos(int k, int n, int KT){
  int tk = k >> 5, tn = n >> 4, kk = k & 31, nn = n & 15;
  int lane = nn | (((kk >> 3) & 1) << 4);
  int j    = ((kk >> 4) << 2) | ((kk >> 1) & 3);
  return ((((size_t)tn * KT + tk) * 32 + lane) << 4) + (j << 1) + (kk & 1);
}

// ---------------- pre-processing kernels ----------------
__global__ void pack_wcat(const float* __restrict__ Wx, const float* __restrict__ Wh,
                          uint16_t* __restrict__ P){
  for (size_t i = (size_t)blockIdx.x * blockDim.x + threadIdx.x;
       i < (size_t)KA * NG; i += (size_t)gridDim.x * blockDim.x){
    int k = (int)(i >> 11), c = (int)(i & 2047);
    float v = (k < Hh) ? Wx[(size_t)k * NG + c] : Wh[(size_t)(k - Hh) * NG + c];
    P[bpos(k, c, 32)] = f2bfu(v);
  }
}
__global__ void pack_wi(const float* __restrict__ Wi, uint16_t* __restrict__ P){
  for (size_t i = (size_t)blockIdx.x * blockDim.x + threadIdx.x;
       i < (size_t)Hh * WINP; i += (size_t)gridDim.x * blockDim.x){
    int k = (int)(i / WINP), c = (int)(i % WINP);
    float v = (c < WIN) ? Wi[(size_t)k * WIN + c] : 0.f;
    P[bpos(k, c, 16)] = f2bfu(v);
  }
}
__global__ void pack_wo(const float* __restrict__ Wo, uint16_t* __restrict__ P){
  // A slice for GEMM3 is [r(256)|h(512)], so reorder Wo rows: r-rows first.
  for (size_t i = (size_t)blockIdx.x * blockDim.x + threadIdx.x;
       i < (size_t)768 * Ww; i += (size_t)gridDim.x * blockDim.x){
    int k = (int)(i >> 8), c = (int)(i & 255);
    int srcRow = (k < Ww) ? (Hh + k) : (k - Ww);
    P[bpos(k, c, 24)] = f2bfu(Wo[(size_t)srcRow * Ww + c]);
  }
}
__global__ void zero_f32(float* __restrict__ p, size_t n){
  for (size_t i = (size_t)blockIdx.x * blockDim.x + threadIdx.x; i < n;
       i += (size_t)gridDim.x * blockDim.x) p[i] = 0.f;
}
__global__ void copy_f32(float* __restrict__ d, const float* __restrict__ s, size_t n){
  for (size_t i = (size_t)blockIdx.x * blockDim.x + threadIdx.x; i < n;
       i += (size_t)gridDim.x * blockDim.x) d[i] = s[i];
}

// ---------------- per-wave softmax over 128 memory slots ----------------
__device__ __forceinline__ void softmax_rows(float* sc, float* outp, int Tlen, int t,
                                             int lane, int wave){
  for (int b = wave; b < Bb; b += 16){
    float vals[4]; float mx = -3.4e38f;
    #pragma unroll
    for (int q = 0; q < 4; ++q){ vals[q] = sc[b * Nm + lane + 32 * q]; mx = fmaxf(mx, vals[q]); }
    #pragma unroll
    for (int d = 16; d > 0; d >>= 1) mx = fmaxf(mx, __shfl_xor(mx, d, 32));
    float s = 0.f;
    #pragma unroll
    for (int q = 0; q < 4; ++q){ vals[q] = __expf(vals[q] - mx); s += vals[q]; }
    #pragma unroll
    for (int d = 16; d > 0; d >>= 1) s += __shfl_xor(s, d, 32);
    float inv = 1.f / s;
    #pragma unroll
    for (int q = 0; q < 4; ++q){
      int n = lane + 32 * q;
      float w = vals[q] * inv;
      sc[b * Nm + n] = w;
      outp[((size_t)b * Tlen + t) * Nm + n] = w;
    }
  }
}

// ---------------- persistent recurrent scan (one workgroup, 16 wave32 waves) ----------------
__global__ __launch_bounds__(512, 1)
void dnc_scan(const int* __restrict__ toks, int Tsteps, int tokStride, int flagIdx, int queryLast,
              const float* __restrict__ emb,
              const uint16_t* __restrict__ WcatP, const uint16_t* __restrict__ WiP,
              const uint16_t* __restrict__ WoP,
              const float* __restrict__ b_lstm, const float* __restrict__ bi,
              const float* __restrict__ bo,
              float* __restrict__ gbuf, float* __restrict__ ifbuf, float* __restrict__ scbuf,
              float* __restrict__ cstate, float* __restrict__ M,
              const uint16_t* hInit, const uint16_t* rInit,
              uint16_t* hOut, uint16_t* rOut,
              float* __restrict__ hrOut, float* __restrict__ hwOut, int Tlen,
              float* __restrict__ aOut){
  __shared__ __align__(16) __bf16 Ash[Bb * LDA];   // A = [xt(256)|r(256)|h(512)] bf16
  __shared__ float KN[Bb];
  __shared__ float BETA[Bb];
  const int tid = threadIdx.x, lane = tid & 31, wave = tid >> 5;

  // init r/h region of A from saved state (or zero)
  for (int i = tid; i < Bb * (Ww + Hh); i += 512){
    int b = i / (Ww + Hh), j = i % (Ww + Hh);
    uint16_t u = 0;
    if (j < Ww) { if (rInit) u = rInit[b * Ww + j]; }
    else        { if (hInit) u = hInit[b * Hh + (j - Ww)]; }
    Ash[b * LDA + WORDW + j] = __builtin_bit_cast(__bf16, u);
  }
  __syncthreads();

  for (int t = 0; t < Tsteps; ++t){
    // ---- P0: embed tokens (+flag) into A cols 0..255 ----
    const bool q = queryLast && (t == Tsteps - 1);
    for (int i = tid; i < Bb * WORDW; i += 512){
      int b = i >> 8, j = i & 255;
      float x = 0.f;
      if (!q){
        int tok = toks[b * tokStride + t];
        x = (j < Dd) ? emb[(size_t)tok * Dd + j] : ((j - Dd) == flagIdx ? 1.f : 0.f);
      }
      Ash[b * LDA + j] = f2bf(x);
    }
    __syncthreads();

    // ---- P1: g = [xt,r,h] @ Wcat + b_lstm  ([64,1024]x[1024,2048]) ----
    {
      int rt = wave & 3, cg = wave >> 2;           // 4 row-tiles x 4 col-groups
      for (int pass = 0; pass < 4; ++pass){
        v8f acc[8];
        #pragma unroll
        for (int i = 0; i < 8; ++i) acc[i] = (v8f){0.f,0.f,0.f,0.f,0.f,0.f,0.f,0.f};
        for (int kt = 0; kt < 32; ++kt){
          v16bf a = loadA(Ash, rt, kt * 32, lane);
          #pragma unroll
          for (int i = 0; i < 8; ++i){
            int ct = (cg * 4 + pass) * 8 + i;
            v16bf b = loadB(WcatP, ct, kt, 32, lane);
            acc[i] = __builtin_amdgcn_wmma_f32_16x16x32_bf16(false, a, false, b,
                                                             (short)0, acc[i], false, false);
          }
        }
        #pragma unroll
        for (int i = 0; i < 8; ++i){
          int ct  = (cg * 4 + pass) * 8 + i;
          int col = ct * 16 + (lane & 15);
          float bia = b_lstm[col];
          #pragma unroll
          for (int v = 0; v < 8; ++v){
            int row = rt * 16 + v + ((lane >> 4) << 3);
            gbuf[row * NG + col] = acc[i][v] + bia;
          }
        }
      }
    }
    __threadfence(); __syncthreads();

    // ---- P2: LSTM gates, update c, h -> A cols 512..1023 ----
    for (int i = tid; i < Bb * Hh; i += 512){
      int b = i >> 9, j = i & 511;
      const float* gr = gbuf + (size_t)b * NG;
      float ii = gr[j], ff = gr[Hh + j], gg = gr[2 * Hh + j], oo = gr[3 * Hh + j];
      float cn = sigf(ff) * cstate[i] + sigf(ii) * tanhf(gg);
      cstate[i] = cn;
      Ash[b * LDA + 512 + j] = f2bf(sigf(oo) * tanhf(cn));
    }
    __syncthreads();

    // ---- P3: iface = h @ Wi + bi  ([64,512]x[512,1040]) ----
    for (int tile = wave; tile < 4 * 65; tile += 16){
      int rt = tile & 3, ct = tile >> 2;
      v8f acc = (v8f){0.f,0.f,0.f,0.f,0.f,0.f,0.f,0.f};
      for (int kt = 0; kt < 16; ++kt){
        v16bf a = loadA(Ash, rt, 512 + kt * 32, lane);
        v16bf b = loadB(WiP, ct, kt, 16, lane);
        acc = __builtin_amdgcn_wmma_f32_16x16x32_bf16(false, a, false, b,
                                                      (short)0, acc, false, false);
      }
      int col = ct * 16 + (lane & 15);
      float bia = (col < WIN) ? bi[col] : 0.f;
      #pragma unroll
      for (int v = 0; v < 8; ++v){
        int row = rt * 16 + v + ((lane >> 4) << 3);
        ifbuf[row * WINP + col] = acc[v] + bia;
      }
    }
    __threadfence(); __syncthreads();

    // ---- P4: write head: ww = softmax(bw * cos(M, kw)); M update ----
    if (tid < Bb){
      const float* kv = ifbuf + (size_t)tid * WINP + Ww;       // kw
      float s = 0.f;
      for (int w = 0; w < Ww; ++w){ float x = kv[w]; s += x * x; }
      KN[tid]   = sqrtf(s);
      BETA[tid] = softplusf(ifbuf[(size_t)tid * WINP + 1025]) + 1.0f;  // bw
    }
    __syncthreads();
    for (int i = tid; i < Bb * Nm; i += 512){
      int b = i >> 7, n = i & 127;
      const float* Mr = M + ((size_t)b * Nm + n) * Ww;
      const float* kv = ifbuf + (size_t)b * WINP + Ww;
      float dot = 0.f, nm = 0.f;
      for (int w = 0; w < Ww; ++w){ float m = Mr[w]; dot += m * kv[w]; nm += m * m; }
      scbuf[i] = BETA[b] * dot / (sqrtf(nm) * KN[b] + EPSV);
    }
    __threadfence(); __syncthreads();
    softmax_rows(scbuf, hwOut, Tlen, t, lane, wave);
    __threadfence(); __syncthreads();
    for (int i = tid; i < Bb * Nm * Ww; i += 512){
      int w = i & 255, n = (i >> 8) & 127, b = i >> 15;
      const float* fb = ifbuf + (size_t)b * WINP;
      float ww = scbuf[b * Nm + n];
      float e  = sigf(fb[3 * Ww + w]);
      float v  = fb[2 * Ww + w];
      M[i] = M[i] * (1.f - ww * e) + ww * v;
    }
    __threadfence(); __syncthreads();

    // ---- P5: read head: wr = softmax(br * cos(M', kr)); r = wr @ M' ----
    if (tid < Bb){
      const float* kv = ifbuf + (size_t)tid * WINP;            // kr
      float s = 0.f;
      for (int w = 0; w < Ww; ++w){ float x = kv[w]; s += x * x; }
      KN[tid]   = sqrtf(s);
      BETA[tid] = softplusf(ifbuf[(size_t)tid * WINP + 1024]) + 1.0f;  // br
    }
    __syncthreads();
    for (int i = tid; i < Bb * Nm; i += 512){
      int b = i >> 7, n = i & 127;
      const float* Mr = M + ((size_t)b * Nm + n) * Ww;
      const float* kv = ifbuf + (size_t)b * WINP;
      float dot = 0.f, nm = 0.f;
      for (int w = 0; w < Ww; ++w){ float m = Mr[w]; dot += m * kv[w]; nm += m * m; }
      scbuf[i] = BETA[b] * dot / (sqrtf(nm) * KN[b] + EPSV);
    }
    __threadfence(); __syncthreads();
    softmax_rows(scbuf, hrOut, Tlen, t, lane, wave);
    __threadfence(); __syncthreads();
    for (int i = tid; i < Bb * Ww; i += 512){
      int b = i >> 8, w = i & 255;
      const float* Mb = M + (size_t)b * Nm * Ww + w;
      const float* wr = scbuf + b * Nm;
      float s = 0.f;
      for (int n = 0; n < Nm; ++n) s += wr[n] * Mb[(size_t)n * Ww];
      Ash[b * LDA + WORDW + w] = f2bf(s);                      // r -> A cols 256..511
    }
    __syncthreads();

    // ---- P6: out = clip([h,r] @ Wo + bo) — only needed at the final option step ----
    if (aOut && t == Tsteps - 1){
      for (int tile = wave; tile < 64; tile += 16){
        int rt = tile & 3, ct = tile >> 2;
        v8f acc = (v8f){0.f,0.f,0.f,0.f,0.f,0.f,0.f,0.f};
        for (int kt = 0; kt < 24; ++kt){
          v16bf a = loadA(Ash, rt, WORDW + kt * 32, lane);     // [r|h] slice
          v16bf b = loadB(WoP, ct, kt, 24, lane);
          acc = __builtin_amdgcn_wmma_f32_16x16x32_bf16(false, a, false, b,
                                                        (short)0, acc, false, false);
        }
        int col = ct * 16 + (lane & 15);
        float bia = bo[col];
        #pragma unroll
        for (int v = 0; v < 8; ++v){
          int row = rt * 16 + v + ((lane >> 4) << 3);
          float x = acc[v] + bia;
          aOut[row * Ww + col] = fminf(fmaxf(x, -CLIPV), CLIPV);
        }
      }
    }
    __syncthreads();
  }

  // ---- save end state (bf16 h/r; c and M already in workspace) ----
  if (hOut){
    for (int i = tid; i < Bb * Hh; i += 512){
      int b = i >> 9, j = i & 511;
      hOut[i] = __builtin_bit_cast(uint16_t, Ash[b * LDA + 512 + j]);
    }
    for (int i = tid; i < Bb * Ww; i += 512){
      int b = i >> 8, j = i & 255;
      rOut[i] = __builtin_bit_cast(uint16_t, Ash[b * LDA + WORDW + j]);
    }
  }
}

// ---------------- final classifier: [a1,a2] @ Wfc + bfc ----------------
__global__ void final_fc(const float* __restrict__ a1, const float* __restrict__ a2,
                         const float* __restrict__ Wfc, const float* __restrict__ bfc,
                         float* __restrict__ out){
  int tid = threadIdx.x;
  if (tid >= 128) return;
  int b = tid >> 1, c = tid & 1;
  float s = bfc[c];
  for (int j = 0; j < 512; ++j){
    float x = (j < 256) ? a1[b * 256 + j] : a2[b * 256 + (j - 256)];
    s += x * Wfc[j * 2 + c];
  }
  out[b * 2 + c] = s;
}

// ---------------- host launcher ----------------
extern "C" void kernel_launch(void* const* d_in, const int* in_sizes, int n_in,
                              void* d_out, int out_size, void* d_ws, size_t ws_size,
                              hipStream_t stream){
  (void)in_sizes; (void)n_in; (void)out_size; (void)ws_size;
  const int*   story  = (const int*)d_in[0];
  const int*   opt1   = (const int*)d_in[1];
  const int*   opt2   = (const int*)d_in[2];
  const float* emb    = (const float*)d_in[3];
  const float* Wx     = (const float*)d_in[4];
  const float* Wh     = (const float*)d_in[5];
  const float* b_lstm = (const float*)d_in[6];
  const float* Wi     = (const float*)d_in[7];
  const float* bi     = (const float*)d_in[8];
  const float* Wo     = (const float*)d_in[9];
  const float* bo     = (const float*)d_in[10];
  const float* Wfc    = (const float*)d_in[11];
  const float* bfc    = (const float*)d_in[12];
  float* out = (float*)d_out;

  char* ws = (char*)d_ws;
  size_t cur = 0;
  auto alloc = [&](size_t bytes) -> void* {
    void* p = ws + cur;
    cur += (bytes + 255) & ~(size_t)255;
    return p;
  };
  uint16_t* WcatP   = (uint16_t*)alloc((size_t)KA * NG * 2);
  uint16_t* WiP     = (uint16_t*)alloc((size_t)Hh * WINP * 2);
  uint16_t* WoP     = (uint16_t*)alloc((size_t)768 * Ww * 2);
  float*    gbuf    = (float*)alloc((size_t)Bb * NG * 4);
  float*    ifbuf   = (float*)alloc((size_t)Bb * WINP * 4);
  float*    scbuf   = (float*)alloc((size_t)Bb * Nm * 4);
  float*    M_story = (float*)alloc((size_t)Bb * Nm * Ww * 4);
  float*    M_work  = (float*)alloc((size_t)Bb * Nm * Ww * 4);
  float*    c_story = (float*)alloc((size_t)Bb * Hh * 4);
  float*    c_work  = (float*)alloc((size_t)Bb * Hh * 4);
  uint16_t* h_end   = (uint16_t*)alloc((size_t)Bb * Hh * 2);
  uint16_t* r_end   = (uint16_t*)alloc((size_t)Bb * Ww * 2);
  float*    a1      = (float*)alloc((size_t)Bb * Ww * 4);
  float*    a2      = (float*)alloc((size_t)Bb * Ww * 4);

  // output layout: out(128), hr_s, hr_1, hr_2, hw_s, hw_1, hw_2
  const size_t o_hrs = 128;
  const size_t o_hr1 = o_hrs + (size_t)Bb * 128 * Nm;
  const size_t o_hr2 = o_hr1 + (size_t)Bb * 17 * Nm;
  const size_t o_hws = o_hr2 + (size_t)Bb * 17 * Nm;
  const size_t o_hw1 = o_hws + (size_t)Bb * 128 * Nm;
  const size_t o_hw2 = o_hw1 + (size_t)Bb * 17 * Nm;

  // pre-pack weights + zero initial state (idempotent each call)
  pack_wcat<<<4096, 256, 0, stream>>>(Wx, Wh, WcatP);
  pack_wi  <<<1024, 256, 0, stream>>>(Wi, WiP);
  pack_wo  <<<512,  256, 0, stream>>>(Wo, WoP);
  zero_f32 <<<2048, 256, 0, stream>>>(M_story, (size_t)Bb * Nm * Ww);
  zero_f32 <<<128,  256, 0, stream>>>(c_story, (size_t)Bb * Hh);

  // story scan: zero init state, emit hr_s / hw_s, save end state
  dnc_scan<<<1, 512, 0, stream>>>(story, 128, 128, 0, 0, emb, WcatP, WiP, WoP,
      b_lstm, bi, bo, gbuf, ifbuf, scbuf, c_story, M_story,
      nullptr, nullptr, h_end, r_end, out + o_hrs, out + o_hws, 128, nullptr);

  // option 1: restore story state, run 17 steps, emit a1
  copy_f32<<<2048, 256, 0, stream>>>(M_work, M_story, (size_t)Bb * Nm * Ww);
  copy_f32<<<128,  256, 0, stream>>>(c_work, c_story, (size_t)Bb * Hh);
  dnc_scan<<<1, 512, 0, stream>>>(opt1, 17, 16, 1, 1, emb, WcatP, WiP, WoP,
      b_lstm, bi, bo, gbuf, ifbuf, scbuf, c_work, M_work,
      h_end, r_end, nullptr, nullptr, out + o_hr1, out + o_hw1, 17, a1);

  // option 2
  copy_f32<<<2048, 256, 0, stream>>>(M_work, M_story, (size_t)Bb * Nm * Ww);
  copy_f32<<<128,  256, 0, stream>>>(c_work, c_story, (size_t)Bb * Hh);
  dnc_scan<<<1, 512, 0, stream>>>(opt2, 17, 16, 2, 1, emb, WcatP, WiP, WoP,
      b_lstm, bi, bo, gbuf, ifbuf, scbuf, c_work, M_work,
      h_end, r_end, nullptr, nullptr, out + o_hr2, out + o_hw2, 17, a2);

  // final classifier
  final_fc<<<1, 128, 0, stream>>>(a1, a2, Wfc, bfc, out);
}
